// TorchGemma4VisionAttention_49331994362292
// MI455X (gfx1250) — matrix operand order repaired and checked by reading, not compiled, not run on hardware
//
#include <hip/hip_runtime.h>

#define NHEADS 16
#define HDIM   72
#define HID    1152
#define SEQ    4096
#define PROJ3  3456

typedef __attribute__((ext_vector_type(16))) __bf16 v16bf;
typedef __attribute__((ext_vector_type(8)))  float  v8f;

union FragU { v16bf v; uint4 q[2]; };

// A-operand (16x32 bf16) from row-major bf16, ld in elements.
// Lane L: row = L&15; lanes 0-15 hold K in [0,16), lanes 16-31 K in [16,32).
__device__ __forceinline__ v16bf load_a_frag(const __bf16* base, int ld, int lane) {
  int m  = lane & 15;
  int kb = (lane & 16) ? 16 : 0;
  const __bf16* p = base + (size_t)m * ld + kb;
  FragU u;
  u.q[0] = *reinterpret_cast<const uint4*>(p);
  u.q[1] = *reinterpret_cast<const uint4*>(p + 8);
  return u.v;
}

// B-operand (32x16 bf16) from pre-packed tile: lane L holds 16 contiguous bf16
// = B[k0 + 16*(L>=16) + j][n = L&15], j=0..15.  Tile = 512 bf16 (1 KB).
__device__ __forceinline__ v16bf load_b_frag(const __bf16* tile, int lane) {
  const __bf16* p = tile + lane * 16;
  FragU u;
  u.q[0] = *reinterpret_cast<const uint4*>(p);
  u.q[1] = *reinterpret_cast<const uint4*>(p + 8);
  return u.v;
}

__device__ __forceinline__ v8f wmma_bf16(v16bf a, v16bf b, v8f c) {
  return __builtin_amdgcn_wmma_f32_16x16x32_bf16(false, a, false, b, (short)0, c,
                                                 false, false);
}

// ---------------- Kernel 1: fp32 -> bf16 convert ----------------
__global__ void k_convert(const float* __restrict__ x, __bf16* __restrict__ y, int n) {
  int i = blockIdx.x * blockDim.x + threadIdx.x;
  if (i < n) y[i] = (__bf16)x[i];
}

// ---------------- Kernel 2: pack weights into WMMA B layout ----------------
// Wq|Wk|Wv fused as one 1152x3456 B matrix (216 n-tiles x 36 k-tiles),
// Wo as 1152x1152 (72 n-tiles x 36 k-tiles). Tile linear index = nt*36 + kt.
__global__ void k_pack_w(const float* __restrict__ Wq, const float* __restrict__ Wk,
                         const float* __restrict__ Wv, const float* __restrict__ Wo,
                         __bf16* __restrict__ wqkv, __bf16* __restrict__ wo) {
  const int QKV_TILES = 216 * 36;
  int tile = blockIdx.x;
  bool isO = tile >= QKV_TILES;
  int t = isO ? (tile - QKV_TILES) : tile;
  int nt = t / 36;
  int kt = t - nt * 36;
  __bf16* dst = (isO ? wo : wqkv) + (size_t)t * 512;
  for (int e = threadIdx.x; e < 512; e += 128) {
    int lane = e >> 4, j = e & 15;
    int k   = kt * 32 + ((lane >> 4) << 4) + j;   // K index
    int col = nt * 16 + (lane & 15);              // N index
    float val;
    if (isO)                 val = Wo[(size_t)k * HID + col];
    else if (col < HID)      val = Wq[(size_t)k * HID + col];
    else if (col < 2 * HID)  val = Wk[(size_t)k * HID + (col - HID)];
    else                     val = Wv[(size_t)k * HID + (col - 2 * HID)];
    dst[e] = (__bf16)val;
  }
}

// ------- Kernel 3/6: bf16 GEMM, wave computes 32x64 C tile, pipelined --------
__global__ void __launch_bounds__(128, 1)
k_gemm(const __bf16* __restrict__ A, int lda,
       const __bf16* __restrict__ Bp, float* __restrict__ C,
       int ldc, int ncChunks) {
  int wave = (blockIdx.x * blockDim.x + threadIdx.x) >> 5;
  int lane = threadIdx.x & 31;
  int mt = wave / ncChunks;          // 32-row tile index
  int nc = wave - mt * ncChunks;     // 64-col chunk index
  int row0 = mt << 5;

  v8f acc00 = {}, acc01 = {}, acc02 = {}, acc03 = {};
  v8f acc10 = {}, acc11 = {}, acc12 = {}, acc13 = {};
  const __bf16* arow0 = A + (size_t)row0 * lda;
  const __bf16* arow1 = A + (size_t)(row0 + 16) * lda;
  const __bf16* b0p = Bp + (size_t)(nc * 4 + 0) * 36 * 512;
  const __bf16* b1p = Bp + (size_t)(nc * 4 + 1) * 36 * 512;
  const __bf16* b2p = Bp + (size_t)(nc * 4 + 2) * 36 * 512;
  const __bf16* b3p = Bp + (size_t)(nc * 4 + 3) * 36 * 512;

  // prologue
  v16bf a0 = load_a_frag(arow0, lda, lane);
  v16bf a1 = load_a_frag(arow1, lda, lane);
  v16bf b0 = load_b_frag(b0p, lane);
  v16bf b1 = load_b_frag(b1p, lane);
  v16bf b2 = load_b_frag(b2p, lane);
  v16bf b3 = load_b_frag(b3p, lane);

#pragma unroll 2
  for (int kt = 0; kt < 36; ++kt) {
    int ktn = (kt + 1 < 36) ? (kt + 1) : 35;   // dup-load last iter, keeps loop branch-free
    // prefetch next iteration's fragments before consuming current ones
    v16bf na0 = load_a_frag(arow0 + ktn * 32, lda, lane);
    v16bf na1 = load_a_frag(arow1 + ktn * 32, lda, lane);
    v16bf nb0 = load_b_frag(b0p + ktn * 512, lane);
    v16bf nb1 = load_b_frag(b1p + ktn * 512, lane);
    v16bf nb2 = load_b_frag(b2p + ktn * 512, lane);
    v16bf nb3 = load_b_frag(b3p + ktn * 512, lane);

    acc00 = wmma_bf16(a0, b0, acc00);
    acc01 = wmma_bf16(a0, b1, acc01);
    acc02 = wmma_bf16(a0, b2, acc02);
    acc03 = wmma_bf16(a0, b3, acc03);
    acc10 = wmma_bf16(a1, b0, acc10);
    acc11 = wmma_bf16(a1, b1, acc11);
    acc12 = wmma_bf16(a1, b2, acc12);
    acc13 = wmma_bf16(a1, b3, acc13);

    a0 = na0; a1 = na1;
    b0 = nb0; b1 = nb1; b2 = nb2; b3 = nb3;
  }

  int r8 = (lane & 16) ? 8 : 0;
  int nl = lane & 15;
  int colBase = nc * 64 + nl;
#pragma unroll
  for (int r = 0; r < 8; ++r) {
    float* crow0 = C + (size_t)(row0 + r + r8) * ldc + colBase;
    crow0[0]  = acc00[r];
    crow0[16] = acc01[r];
    crow0[32] = acc02[r];
    crow0[48] = acc03[r];
    float* crow1 = C + (size_t)(row0 + 16 + r + r8) * ldc + colBase;
    crow1[0]  = acc10[r];
    crow1[16] = acc11[r];
    crow1[32] = acc12[r];
    crow1[48] = acc13[r];
  }
}

// ---------------- Kernel 4: RMS-norm + 2D RoPE + pack Q/K/V ----------------
// grid (SEQ, NHEADS), block 96.  Q -> A-layout row-major padded to 96,
// K -> B tiles (k=dim 96, n=key), V -> B tiles (k=key, n=dim padded to 80).
__global__ void k_norm_rope(const float* __restrict__ XQKV,
                            const float* __restrict__ cosb, const float* __restrict__ sinb,
                            const float* __restrict__ qsc, const float* __restrict__ ksc,
                            __bf16* __restrict__ Qpad, __bf16* __restrict__ Kpack,
                            __bf16* __restrict__ Vpack) {
  int n = blockIdx.x, h = blockIdx.y, d = threadIdx.x;   // d in [0,96)
  __shared__ float shq[96], shk[96], shv[96];
  __shared__ float tot[3];

  const float* base = XQKV + (size_t)n * PROJ3 + h * HDIM;
  float q = 0.f, k = 0.f, v = 0.f;
  if (d < HDIM) { q = base[d]; k = base[HID + d]; v = base[2 * HID + d]; }
  shq[d] = q * q; shk[d] = k * k; shv[d] = v * v;
  __syncthreads();

  if (d < 32) {
    float aq = shq[d] + shq[d + 32] + shq[d + 64];
    float ak = shk[d] + shk[d + 32] + shk[d + 64];
    float av = shv[d] + shv[d + 32] + shv[d + 64];
#pragma unroll
    for (int off = 16; off >= 1; off >>= 1) {
      aq += __shfl_xor(aq, off, 32);
      ak += __shfl_xor(ak, off, 32);
      av += __shfl_xor(av, off, 32);
    }
    if (d == 0) { tot[0] = aq; tot[1] = ak; tot[2] = av; }
  }
  __syncthreads();

  float rq = rsqrtf(tot[0] * (1.f / HDIM) + 1e-6f);
  float rk = rsqrtf(tot[1] * (1.f / HDIM) + 1e-6f);
  float rv = rsqrtf(tot[2] * (1.f / HDIM) + 1e-6f);
  float qn = 0.f, kn = 0.f, vn = 0.f;
  if (d < HDIM) { qn = q * rq * qsc[d]; kn = k * rk * ksc[d]; vn = v * rv; }
  __syncthreads();
  shq[d] = qn; shk[d] = kn;
  __syncthreads();

  float qr = 0.f, kr = 0.f;
  if (d < HDIM) {
    int half = (d < 36) ? 0 : 1;
    int i = d - half * 36;
    float c = cosb[(size_t)n * HDIM + d];
    float s = sinb[(size_t)n * HDIM + d];
    int rot = half * 36 + ((i < 18) ? (i + 18) : (i - 18));
    float sgn = (i < 18) ? -1.f : 1.f;
    qr = shq[d] * c + sgn * shq[rot] * s;
    kr = shk[d] * c + sgn * shk[rot] * s;
  }

  // Q: row-major [head][n][96]
  Qpad[((size_t)h * SEQ + n) * 96 + d] = (__bf16)qr;

  // K: B tiles [head][keytile16][kstep3][lane32][16]
  {
    int kt = d >> 5, rem = d & 31;
    int lane = ((rem >> 4) << 4) | (n & 15);
    int j = rem & 15;
    size_t idx = ((((size_t)h * 256 + (n >> 4)) * 3 + kt) * 512) + lane * 16 + j;
    Kpack[idx] = (__bf16)kr;
  }

  // V: B tiles [head][keytile32][ntile5][lane32][16]  (dim padded to 80)
  if (d < 80) {
    int ntile = d >> 4;
    int rem = n & 31;
    int lane = ((rem >> 4) << 4) | (d & 15);
    int j = rem & 15;
    size_t idx = ((((size_t)h * 128 + (n >> 5)) * 5 + ntile) * 512) + lane * 16 + j;
    Vpack[idx] = (__bf16)vn;
  }
}

// ---------------- Kernel 5: flash attention, 1 wave per (head, 16 queries) ----
__global__ void __launch_bounds__(32, 1)
k_attn(const __bf16* __restrict__ Qpad, const __bf16* __restrict__ Kp,
       const __bf16* __restrict__ Vp, __bf16* __restrict__ AO) {
  int qt = blockIdx.x;
  int h  = blockIdx.y;
  int lane = threadIdx.x;
  __shared__ __align__(16) __bf16 pl[16 * 32];

  const __bf16* qb = Qpad + ((size_t)h * SEQ + qt * 16) * 96;
  v16bf a0 = load_a_frag(qb,      96, lane);
  v16bf a1 = load_a_frag(qb + 32, 96, lane);
  v16bf a2 = load_a_frag(qb + 64, 96, lane);

  v8f o0 = {}, o1 = {}, o2 = {}, o3 = {}, o4 = {};
  float rmax[8], rsum[8];
#pragma unroll
  for (int r = 0; r < 8; ++r) { rmax[r] = -1e30f; rsum[r] = 0.f; }

  const __bf16* Kb = Kp + (size_t)h * 256 * 3 * 512;
  const __bf16* Vb = Vp + (size_t)h * 128 * 5 * 512;
  int r8 = (lane & 16) ? 8 : 0;
  int nl = lane & 15;

  for (int t = 0; t < 128; ++t) {
    const __bf16* kt0 = Kb + (size_t)(2 * t)     * 3 * 512;
    const __bf16* kt1 = Kb + (size_t)(2 * t + 1) * 3 * 512;
    const __bf16* vt  = Vb + (size_t)t * 5 * 512;

    // issue all loads for this iteration up front: 6 K frags + 5 V frags.
    v16bf bk00 = load_b_frag(kt0,        lane);
    v16bf bk01 = load_b_frag(kt0 + 512,  lane);
    v16bf bk02 = load_b_frag(kt0 + 1024, lane);
    v16bf bk10 = load_b_frag(kt1,        lane);
    v16bf bk11 = load_b_frag(kt1 + 512,  lane);
    v16bf bk12 = load_b_frag(kt1 + 1024, lane);
    v16bf bv0  = load_b_frag(vt,         lane);
    v16bf bv1  = load_b_frag(vt + 512,   lane);
    v16bf bv2  = load_b_frag(vt + 1024,  lane);
    v16bf bv3  = load_b_frag(vt + 1536,  lane);
    v16bf bv4  = load_b_frag(vt + 2048,  lane);

    v8f s0 = {}, s1 = {};
    s0 = wmma_bf16(a0, bk00, s0);
    s0 = wmma_bf16(a1, bk01, s0);
    s0 = wmma_bf16(a2, bk02, s0);
    s1 = wmma_bf16(a0, bk10, s1);
    s1 = wmma_bf16(a1, bk11, s1);
    s1 = wmma_bf16(a2, bk12, s1);

    float scale[8];
#pragma unroll
    for (int r = 0; r < 8; ++r) {
      float mx = fmaxf(s0[r], s1[r]);
      mx = fmaxf(mx, __shfl_xor(mx, 1, 32));
      mx = fmaxf(mx, __shfl_xor(mx, 2, 32));
      mx = fmaxf(mx, __shfl_xor(mx, 4, 32));
      mx = fmaxf(mx, __shfl_xor(mx, 8, 32));
      float nm = fmaxf(rmax[r], mx);
      scale[r] = __expf(rmax[r] - nm);
      rmax[r] = nm;
      float p0 = __expf(s0[r] - nm);
      float p1 = __expf(s1[r] - nm);
      s0[r] = p0; s1[r] = p1;
      float sm = p0 + p1;
      sm += __shfl_xor(sm, 1, 32);
      sm += __shfl_xor(sm, 2, 32);
      sm += __shfl_xor(sm, 4, 32);
      sm += __shfl_xor(sm, 8, 32);
      rsum[r] = rsum[r] * scale[r] + sm;
    }
#pragma unroll
    for (int r = 0; r < 8; ++r) {
      o0[r] *= scale[r]; o1[r] *= scale[r]; o2[r] *= scale[r];
      o3[r] *= scale[r]; o4[r] *= scale[r];
    }

    // P (C-layout) -> LDS row-major 16x32 -> reload as A-operand
#pragma unroll
    for (int r = 0; r < 8; ++r) {
      pl[(r + r8) * 32 + nl]      = (__bf16)s0[r];
      pl[(r + r8) * 32 + 16 + nl] = (__bf16)s1[r];
    }
    asm volatile("s_wait_dscnt 0" ::: "memory");
    v16bf pa = load_a_frag(pl, 32, lane);

    o0 = wmma_bf16(pa, bv0, o0);
    o1 = wmma_bf16(pa, bv1, o1);
    o2 = wmma_bf16(pa, bv2, o2);
    o3 = wmma_bf16(pa, bv3, o3);
    o4 = wmma_bf16(pa, bv4, o4);
  }

#pragma unroll
  for (int r = 0; r < 8; ++r) {
    float inv = 1.f / rsum[r];
    __bf16* orow = AO + (size_t)(qt * 16 + r + r8) * HID + h * HDIM;
    orow[nl]      = (__bf16)(o0[r] * inv);
    orow[16 + nl] = (__bf16)(o1[r] * inv);
    orow[32 + nl] = (__bf16)(o2[r] * inv);
    orow[48 + nl] = (__bf16)(o3[r] * inv);
    if (nl < 8) orow[64 + nl] = (__bf16)(o4[r] * inv);
  }
}

// ---------------- host ----------------
extern "C" void kernel_launch(void* const* d_in, const int* in_sizes, int n_in,
                              void* d_out, int out_size, void* d_ws, size_t ws_size,
                              hipStream_t stream) {
  (void)in_sizes; (void)n_in; (void)out_size; (void)ws_size;
  const float* hs   = (const float*)d_in[0];
  const float* cosb = (const float*)d_in[1];
  const float* sinb = (const float*)d_in[2];
  const float* Wq   = (const float*)d_in[3];
  const float* Wk   = (const float*)d_in[4];
  const float* Wv   = (const float*)d_in[5];
  const float* Wo   = (const float*)d_in[6];
  const float* qsc  = (const float*)d_in[7];
  const float* ksc  = (const float*)d_in[8];
  float* out = (float*)d_out;
  char* ws = (char*)d_ws;

  constexpr size_t offX    = 0;                                        // X bf16
  constexpr size_t offWqkv = offX    + (size_t)SEQ * HID * 2;          // 9.4 MB
  constexpr size_t offWo   = offWqkv + (size_t)216 * 36 * 512 * 2;     // 8.0 MB
  constexpr size_t offQ    = offWo   + (size_t)72 * 36 * 512 * 2;      // 2.7 MB
  constexpr size_t offK    = offQ    + (size_t)NHEADS * SEQ * 96 * 2;  // 12.6 MB
  constexpr size_t offV    = offK    + (size_t)NHEADS * 256 * 3 * 512 * 2; // 12.6 MB
  constexpr size_t offXQKV = offV    + (size_t)NHEADS * 128 * 5 * 512 * 2; // 10.5 MB
  constexpr size_t offAO   = offXQKV;  // reuse QKV fp32 region after attention inputs packed

  __bf16* Xb    = (__bf16*)(ws + offX);
  __bf16* Wqkvp = (__bf16*)(ws + offWqkv);
  __bf16* Wop   = (__bf16*)(ws + offWo);
  __bf16* Qpad  = (__bf16*)(ws + offQ);
  __bf16* Kp    = (__bf16*)(ws + offK);
  __bf16* Vp    = (__bf16*)(ws + offV);
  float*  XQKV  = (float*)(ws + offXQKV);
  __bf16* AO    = (__bf16*)(ws + offAO);

  k_convert<<<(SEQ * HID + 255) / 256, 256, 0, stream>>>(hs, Xb, SEQ * HID);
  k_pack_w<<<216 * 36 + 72 * 36, 128, 0, stream>>>(Wq, Wk, Wv, Wo, Wqkvp, Wop);
  // QKV fused GEMM: C = X(4096x1152) * Wqkv(1152x3456), 32x64 per wave
  k_gemm<<<(128 * 54) / 4, 128, 0, stream>>>(Xb, HID, Wqkvp, XQKV, PROJ3, 54);
  k_norm_rope<<<dim3(SEQ, NHEADS), 96, 0, stream>>>(XQKV, cosb, sinb, qsc, ksc,
                                                    Qpad, Kp, Vp);
  k_attn<<<dim3(SEQ / 16, NHEADS), 32, 0, stream>>>(Qpad, Kp, Vp, AO);
  // Output GEMM: out = AO(4096x1152) * Wo(1152x1152), 32x64 per wave
  k_gemm<<<(128 * 18) / 4, 128, 0, stream>>>(AO, HID, Wop, out, HID, 18);
}
